// EnhancedGATModel_16862041604781
// MI455X (gfx1250) — matrix operand order repaired and checked by reading, not compiled, not run on hardware
//
#include <hip/hip_runtime.h>
#include <hip/hip_bf16.h>
#include <math.h>

// ---------------------------------------------------------------------------
// Types for CDNA5 WMMA (gfx1250, wave32)
// ---------------------------------------------------------------------------
typedef __bf16 bf16;
typedef __attribute__((ext_vector_type(16))) bf16  v16bf;
typedef __attribute__((ext_vector_type(8)))  bf16  v8bf;
typedef __attribute__((ext_vector_type(8)))  float v8f;

#define HID 128
#define FEAT 64

// ---------------------------------------------------------------------------
// Helpers
// ---------------------------------------------------------------------------
__device__ __forceinline__ float lrelu(float x) { return x > 0.f ? x : 0.2f * x; }

__device__ __forceinline__ unsigned pack2bf(float a, float b) {
    union { bf16 h[2]; unsigned u; } p;
    p.h[0] = (bf16)a; p.h[1] = (bf16)b;
    return p.u;
}

__device__ __forceinline__ v16bf cat8(v8bf lo, v8bf hi) {
    return __builtin_shufflevector(lo, hi, 0, 1, 2, 3, 4, 5, 6, 7,
                                           8, 9, 10, 11, 12, 13, 14, 15);
}

// monotonic float <-> uint encoding so atomicMax(unsigned) == float max
__device__ __forceinline__ unsigned fenc(float f) {
    unsigned u = __float_as_uint(f);
    return (u & 0x80000000u) ? ~u : (u | 0x80000000u);
}
__device__ __forceinline__ float fdec(unsigned u) {
    unsigned b = (u & 0x80000000u) ? (u & 0x7FFFFFFFu) : ~u;
    return __uint_as_float(b);
}

// ---------------------------------------------------------------------------
// Small utility kernels
// ---------------------------------------------------------------------------
__global__ void k_zero(float* __restrict__ p, long long n) {
    long long i = (long long)blockIdx.x * blockDim.x + threadIdx.x;
    long long stride = (long long)gridDim.x * blockDim.x;
    for (; i < n; i += stride) p[i] = 0.f;
}

// v[k] = sum_h W[k*128+h] * a[h]   (W is [K,128] row-major)
__global__ void k_matvec_wa(const float* __restrict__ W, const float* __restrict__ a,
                            float* __restrict__ v, int K) {
    int k = threadIdx.x;
    if (k < K) {
        float s = 0.f;
        #pragma unroll 8
        for (int h = 0; h < HID; ++h) s += W[k * HID + h] * a[h];
        v[k] = s;
    }
}

// out[i] = dot(X[i, 0:K], v)   (attention logit term via (XW)a = X(Wa))
__global__ void k_rowdot(const float* __restrict__ X, int ldx,
                         const float* __restrict__ v, float* __restrict__ out,
                         int M, int K) {
    int i = blockIdx.x * blockDim.x + threadIdx.x;
    if (i >= M) return;
    const float* row = X + (size_t)i * ldx;
    float s = 0.f;
    for (int k = 0; k < K; ++k) s += row[k] * v[k];
    out[i] = s;
}

// init segment-max to enc(-inf) and segment-sum to 0
__global__ void k_init_seg(unsigned* __restrict__ m, float* __restrict__ s, int N) {
    int i = blockIdx.x * blockDim.x + threadIdx.x;
    if (i < N) { m[i] = 0x007FFFFFu; /* fenc(-inf) */ s[i] = 0.f; }
}

__global__ void k_edge_max(const int* __restrict__ src, const int* __restrict__ dst,
                           const float* __restrict__ es, const float* __restrict__ ed,
                           unsigned* __restrict__ m, int E) {
    int e = blockIdx.x * blockDim.x + threadIdx.x;
    if (e >= E) return;
    int d = dst[e];
    float x = lrelu(es[src[e]] + ed[d]);
    atomicMax(&m[d], fenc(x));
}

__global__ void k_edge_exp(const int* __restrict__ src, const int* __restrict__ dst,
                           const float* __restrict__ es, const float* __restrict__ ed,
                           const unsigned* __restrict__ m, float* __restrict__ ee,
                           float* __restrict__ ss, int E) {
    int e = blockIdx.x * blockDim.x + threadIdx.x;
    if (e >= E) return;
    int d = dst[e];
    float x = lrelu(es[src[e]] + ed[d]);
    float v = expf(x - fdec(m[d]));
    ee[e] = v;
    atomicAdd(&ss[d], v);
}

// acc[dst, h] += (ee[e]/ss[dst]) * hs[src, h]  -- one thread per (edge, h)
__global__ void k_edge_scatter(const int* __restrict__ src, const int* __restrict__ dst,
                               const float* __restrict__ ee, const float* __restrict__ ss,
                               const float* __restrict__ hs, float* __restrict__ acc,
                               long long total) {
    long long idx = (long long)blockIdx.x * blockDim.x + threadIdx.x;
    long long stride = (long long)gridDim.x * blockDim.x;
    for (; idx < total; idx += stride) {
        int e = (int)(idx >> 7);
        int h = (int)(idx & 127);
        int sN = src[e], dN = dst[e];
        float alpha = ee[e] / (ss[dN] + 1e-16f);
        atomicAdd(&acc[(size_t)dN * HID + h], alpha * hs[(size_t)sN * HID + h]);
    }
}

__global__ void k_bias_add(float* __restrict__ acc, const float* __restrict__ bias,
                           long long total) {
    long long idx = (long long)blockIdx.x * blockDim.x + threadIdx.x;
    if (idx < total) acc[idx] += bias[idx & 127];
}

__global__ void k_finalize(float* __restrict__ hout, const float* __restrict__ acc,
                           float invCnt, long long total) {
    long long idx = (long long)blockIdx.x * blockDim.x + threadIdx.x;
    if (idx < total) hout[idx] = tanhf(acc[idx] * invCnt);
}

// ---------------------------------------------------------------------------
// GEMM: H[M,128] = X[M,K] @ W[K,128], bf16 WMMA, fp32 accumulate.
// Block = 128 threads (4 waves). Each block: 64 rows; wave w -> rows w*16..+16,
// all 8 N-tiles of 16. A staged row-major, W staged TRANSPOSED so every
// fragment is two 16B-contiguous chunks -> ds_load_b128 pairs.
// Fragment layouts per CDNA5 ISA 7.12.2 (wave32):
//   A 16x32: lane l -> M=l%16, half=l/16; elem i -> K=(i%8)+8*half+16*(i/8)
//            => chunks [kt*32+8*half, +8) and [kt*32+16+8*half, +8)
//   B 32x16: lane l -> N=l%16;           elem i -> K=(i%8)+8*half+16*(i/8)
//            => chunks [kt*32+8*half, +8) and [kt*32+16+8*half, +8) at fixed N (transposed layout)
//   C 16x16: lane l -> N=l%16; vgpr j -> M=j+8*(l/16)
// ---------------------------------------------------------------------------
__global__ void k_gemm_n128(const float* __restrict__ X, int ldx,
                            const float* __restrict__ W,
                            float* __restrict__ H, int M, int K) {
    __shared__ bf16 Alds[64][136];   // rows x K (K<=128); stride 136 (16B-aligned rows)
    __shared__ bf16 Wt[128][136];    // [n][k] transposed

    const int tid = threadIdx.x;
    const int m0 = blockIdx.x * 64;
    const int Kp = K >> 1;

    // stage A, packing 2 bf16 per 32-bit LDS store
    for (int t = tid; t < 64 * Kp; t += blockDim.x) {
        int r = t / Kp, c = (t % Kp) * 2;
        int row = m0 + r;
        float x0 = 0.f, x1 = 0.f;
        if (row < M) {
            const float* xp = X + (size_t)row * ldx + c;
            x0 = xp[0]; x1 = xp[1];
        }
        *(unsigned*)&Alds[r][c] = pack2bf(x0, x1);
    }
    // stage W transposed: Wt[n][k] = W[k][n]; global reads coalesced over n
    for (int t = tid; t < Kp * HID; t += blockDim.x) {
        int kp = t / HID, n = t % HID;
        int k = kp * 2;
        *(unsigned*)&Wt[n][k] = pack2bf(W[(size_t)k * HID + n],
                                        W[(size_t)(k + 1) * HID + n]);
    }
    __syncthreads();

    const int wave = tid >> 5;
    const int lane = tid & 31;
    const int rbase = wave * 16;
    const int mA = lane & 15;
    const int half = lane >> 4;

    v8f acc[8] = {};

    const int kSteps = K >> 5;
    for (int kt = 0; kt < kSteps; ++kt) {
        const int ka = kt * 32 + 8 * half;
        v8bf a0 = *(const v8bf*)&Alds[rbase + mA][ka];
        v8bf a1 = *(const v8bf*)&Alds[rbase + mA][ka + 16];
        v16bf a = cat8(a0, a1);

        const int kb = kt * 32 + 8 * half;
        #pragma unroll
        for (int nt = 0; nt < 8; ++nt) {
            int n = nt * 16 + (lane & 15);
            v8bf b0 = *(const v8bf*)&Wt[n][kb];
            v8bf b1 = *(const v8bf*)&Wt[n][kb + 16];
            v16bf b = cat8(b0, b1);
            acc[nt] = __builtin_amdgcn_wmma_f32_16x16x32_bf16(
                false, a, false, b, (short)0, acc[nt], false, false);
        }
    }

    #pragma unroll
    for (int nt = 0; nt < 8; ++nt) {
        #pragma unroll
        for (int j = 0; j < 8; ++j) {
            int row = m0 + rbase + j + 8 * half;
            int col = nt * 16 + (lane & 15);
            if (row < M) H[(size_t)row * HID + col] = acc[nt][j];
        }
    }
}

// ---------------------------------------------------------------------------
// Fused head: combined[M,384] = [pending | user | uh]; z = tanh(combined@W1+b1);
// out = z@W2+b2 (W2 is 768x2). Block = 128 threads, 16 rows/block.
// W1 staged per 32-K slab TRANSPOSED (Wt[768][40] bf16, 61.4KB), aliased with
// the 16x768 f32 z tile. A tile 16x384 bf16 row-major (stride 392).
// ---------------------------------------------------------------------------
__global__ void k_mlp(const float* __restrict__ hp, const float* __restrict__ hu,
                      const float* __restrict__ hh,
                      const float* __restrict__ w1, const float* __restrict__ b1,
                      const float* __restrict__ w2, const float* __restrict__ b2,
                      float* __restrict__ out, int M) {
    __shared__ bf16 Alds[16][392];                     // 16 x 384 (+pad), 12.5KB
    __shared__ __align__(16) char smem[768 * 40 * 2];  // 61.4KB: Wt slab / z tile
    bf16 (*Wt)[40] = (bf16(*)[40])smem;                // [n][k], row = 80B (16B mult.)
    float (*zlds)[768] = (float(*)[768])smem;

    const int tid = threadIdx.x;
    const int m0 = blockIdx.x * 16;
    const int wave = tid >> 5;
    const int lane = tid & 31;
    const int half = lane >> 4;
    const int cb = wave * 192;   // 12 tiles of 16 cols per wave -> 768 total

    // stage combined A tile (concat of three 128-wide blocks), packed bf16 pairs
    for (int t = tid; t < 16 * 192; t += blockDim.x) {
        int r = t / 192, c = (t % 192) * 2;
        int row = m0 + r;
        float v0 = 0.f, v1 = 0.f;
        if (row < M) {
            const float* srcp;
            if (c < 128)       srcp = hp + (size_t)row * HID + c;
            else if (c < 256)  srcp = hu + (size_t)row * HID + (c - 128);
            else               srcp = hh + (size_t)row * HID + (c - 256);
            v0 = srcp[0]; v1 = srcp[1];
        }
        *(unsigned*)&Alds[r][c] = pack2bf(v0, v1);
    }

    v8f acc[12] = {};

    for (int kt = 0; kt < 12; ++kt) {
        __syncthreads();
        // stage 32x768 W1 slab transposed; global reads coalesced over n
        for (int t = tid; t < 16 * 768; t += blockDim.x) {
            int n = t % 768, kp = t / 768;
            int k = kp * 2;
            *(unsigned*)&Wt[n][k] = pack2bf(w1[(size_t)(kt * 32 + k) * 768 + n],
                                            w1[(size_t)(kt * 32 + k + 1) * 768 + n]);
        }
        __syncthreads();

        const int ka = kt * 32 + 8 * half;
        v8bf a0 = *(const v8bf*)&Alds[lane & 15][ka];
        v8bf a1 = *(const v8bf*)&Alds[lane & 15][ka + 16];
        v16bf a = cat8(a0, a1);

        const int kb = 8 * half;
        #pragma unroll
        for (int nt = 0; nt < 12; ++nt) {
            int n = cb + nt * 16 + (lane & 15);
            v8bf b0 = *(const v8bf*)&Wt[n][kb];
            v8bf b1 = *(const v8bf*)&Wt[n][kb + 16];
            v16bf b = cat8(b0, b1);
            acc[nt] = __builtin_amdgcn_wmma_f32_16x16x32_bf16(
                false, a, false, b, (short)0, acc[nt], false, false);
        }
    }
    __syncthreads();  // done reading Wt; smem becomes z tile

    #pragma unroll
    for (int nt = 0; nt < 12; ++nt) {
        int col = cb + nt * 16 + (lane & 15);
        #pragma unroll
        for (int j = 0; j < 8; ++j) {
            int r = j + 8 * half;
            zlds[r][col] = tanhf(acc[nt][j] + b1[col]);
        }
    }
    __syncthreads();

    // skinny lin2: 32 threads = (16 rows) x (2 outputs)
    if (tid < 32) {
        int r = tid >> 1, o = tid & 1;
        int row = m0 + r;
        if (row < M) {
            float s = b2[o];
            for (int c = 0; c < 768; ++c) s += zlds[r][c] * w2[c * 2 + o];
            out[(size_t)row * 2 + o] = s;
        }
    }
}

// ---------------------------------------------------------------------------
// Host-side graph description (mirrors the reference)
// ---------------------------------------------------------------------------
// node types: 0 user, 1 card, 2 user_history, 3 uh_transaction, 4 merchant, 5 pending
static const int       NSZ[6]  = {100000, 150000, 100000, 400000, 20000, 100000};
static const long long NOFF[6] = {0, 100000, 250000, 350000, 750000, 770000};
static const float     INVCNT[6] = {0.5f, 0.5f, 0.5f, 0.5f, 1.0f, 1.0f / 3.0f};
struct EdgeT { int s, d, E; };
static const EdgeT ET[12] = {
    {0, 1, 150000}, {1, 0, 150000}, {0, 2, 100000}, {2, 0, 100000},
    {3, 2, 400000}, {3, 1, 400000}, {3, 4, 400000}, {1, 3, 400000},
    {4, 3, 400000}, {2, 5, 100000}, {4, 5, 100000}, {0, 5, 100000},
};
static const long long TOTN = 870000;

static inline int gsz(long long n, int b) { return (int)((n + b - 1) / b); }

extern "C" void kernel_launch(void* const* d_in, const int* in_sizes, int n_in,
                              void* d_out, int out_size, void* d_ws, size_t ws_size,
                              hipStream_t stream) {
    (void)in_sizes; (void)n_in; (void)out_size; (void)ws_size;

    // -------- workspace partition (floats) --------
    float* ws = (float*)d_ws;
    float* h0 = ws;                                    // 870000*128
    float* h1 = h0 + (size_t)TOTN * HID;               // 870000*128 (accumulator)
    float* hs = h1 + (size_t)TOTN * HID;               // 400000*128 (per-edge-type src proj)
    float* es = hs + (size_t)400000 * HID;             // 400000
    float* ed = es + 400000;                           // 400000
    float* mx = ed + 400000;                           // 400000 (encoded max)
    float* ss = mx + 400000;                           // 400000 (softmax denom)
    float* ee = ss + 400000;                           // 400000 (per-edge exp)
    float* vv = ee + 400000;                           // 256 (W@a vectors)

    for (int li = 0; li < 2; ++li) {
        const int K = (li == 0) ? FEAT : HID;

        // zero dst accumulators for this layer
        {
            long long n = TOTN * HID;
            k_zero<<<gsz(n, 256), 256, 0, stream>>>(h1, n);
        }

        for (int t = 0; t < 12; ++t) {
            const int s = ET[t].s, d = ET[t].d, E = ET[t].E;
            const int Ns = NSZ[s], Nd = NSZ[d];

            const float* Xs = (li == 0) ? (const float*)d_in[s] : h0 + NOFF[s] * HID;
            const float* Xd = (li == 0) ? (const float*)d_in[d] : h0 + NOFF[d] * HID;
            const int ldx = K;

            const int pb = 18 + li * 60 + t * 5;
            const float* Wsrc = (const float*)d_in[pb + 0];
            const float* Wdst = (const float*)d_in[pb + 1];
            const float* asrc = (const float*)d_in[pb + 2];
            const float* adst = (const float*)d_in[pb + 3];
            const float* bias = (const float*)d_in[pb + 4];

            const int* edge = (const int*)d_in[6 + t];
            const int* src = edge;
            const int* dst = edge + E;

            float* accD = h1 + NOFF[d] * HID;

            // attention logit terms via (X W) a == X (W a)  -- exact algebra
            k_matvec_wa<<<1, 128, 0, stream>>>(Wsrc, asrc, vv, K);
            k_matvec_wa<<<1, 128, 0, stream>>>(Wdst, adst, vv + 128, K);
            k_rowdot<<<gsz(Ns, 256), 256, 0, stream>>>(Xs, ldx, vv, es, Ns, K);
            k_rowdot<<<gsz(Nd, 256), 256, 0, stream>>>(Xd, ldx, vv + 128, ed, Nd, K);

            // source projection (the only big GEMM) via bf16 WMMA
            k_gemm_n128<<<gsz(Ns, 64), 128, 0, stream>>>(Xs, ldx, Wsrc, hs, Ns, K);

            // segment softmax over dst + weighted scatter
            k_init_seg<<<gsz(Nd, 256), 256, 0, stream>>>((unsigned*)mx, ss, Nd);
            k_edge_max<<<gsz(E, 256), 256, 0, stream>>>(src, dst, es, ed, (unsigned*)mx, E);
            k_edge_exp<<<gsz(E, 256), 256, 0, stream>>>(src, dst, es, ed,
                                                        (const unsigned*)mx, ee, ss, E);
            {
                long long tot = (long long)E * HID;
                int blocks = gsz(tot, 256);
                if (blocks > 262144) blocks = 262144;  // grid-stride
                k_edge_scatter<<<blocks, 256, 0, stream>>>(src, dst, ee, ss, hs, accD, tot);
            }
            k_bias_add<<<gsz((long long)Nd * HID, 256), 256, 0, stream>>>(
                accD, bias, (long long)Nd * HID);
        }

        // h0 <- tanh(acc / cnt) per node type
        for (int nt = 0; nt < 6; ++nt) {
            long long tot = (long long)NSZ[nt] * HID;
            k_finalize<<<gsz(tot, 256), 256, 0, stream>>>(
                h0 + NOFF[nt] * HID, h1 + NOFF[nt] * HID, INVCNT[nt], tot);
        }
    }

    // fused MLP head on [pending | user | user_history]
    const float* w1 = (const float*)d_in[138];
    const float* b1 = (const float*)d_in[139];
    const float* w2 = (const float*)d_in[140];
    const float* b2 = (const float*)d_in[141];
    k_mlp<<<gsz(100000, 16), 128, 0, stream>>>(
        h0 + NOFF[5] * HID, h0 + NOFF[0] * HID, h0 + NOFF[2] * HID,
        w1, b1, w2, b2, (float*)d_out, 100000);
}
